// GAT_13967233647399
// MI455X (gfx1250) — matrix operand order repaired and checked by reading, not compiled, not run on hardware
//
#include <hip/hip_runtime.h>
#include <hip/hip_bf16.h>
#include <math.h>

// ---------------- Problem constants ----------------
#define NN      50000
#define IN_DIM  128
#define HID     32
#define HEADS   8
#define OUT_DIM 40
#define EE      400000
#define EA      (EE + NN)          // edges + self loops
#define F1      (HEADS * HID)      // 256
#define NEG_SLOPE 0.2f

typedef __attribute__((ext_vector_type(16))) __bf16 v16bf;
typedef __attribute__((ext_vector_type(8)))  __bf16 v8bf;
typedef __attribute__((ext_vector_type(8)))  float  v8f;
typedef __attribute__((ext_vector_type(4)))  float  v4f;

static __device__ __forceinline__ float atomicMaxFloat(float* addr, float value) {
    // correct for mixed signs when initialized to -inf
    if (value >= 0.0f)
        return __int_as_float(atomicMax((int*)addr, __float_as_int(value)));
    else
        return __uint_as_float(atomicMin((unsigned int*)addr, __float_as_uint(value)));
}

// ---------------- Utility kernels ----------------
__global__ void fill_f32(float* p, float v, int n) {
    int i = blockIdx.x * blockDim.x + threadIdx.x;
    if (i < n) p[i] = v;
}

// Pack f32 weight matrix W[K,Nc] into per-lane WMMA B-fragment layout (bf16):
// Bpk[(((tn*(K/32)+kb)*32)+lane)*16 + e] = W[(kb*32 + (lane>>4)*16 + e)*Nc + tn*16 + (lane&15)]
__global__ void pack_b_frag(const float* __restrict__ W, __bf16* __restrict__ Bpk,
                            int K, int Nc) {
    int t = blockIdx.x * blockDim.x + threadIdx.x;
    int kblocks = K >> 5;
    int total = (Nc >> 4) * kblocks * 32;
    if (t >= total) return;
    int lane = t & 31;
    int kb   = (t >> 5) % kblocks;
    int tn   = (t >> 5) / kblocks;
    int col   = lane & 15;
    int kbase = (kb << 5) + ((lane >> 4) << 4);
    __bf16* dst = Bpk + (size_t)t * 16;
    #pragma unroll
    for (int e = 0; e < 16; ++e)
        dst[e] = (__bf16)W[(size_t)(kbase + e) * Nc + (tn << 4) + col];
}

// ---------------- WMMA bf16 GEMM ----------------
// Out[M,Nc] (bf16) = A[M,K] (f32, cvt on load) @ B (pre-packed bf16 fragments)
// One wave computes NT adjacent 16x16 N-tiles of one M-tile (A fragment reused).
template<int K, int NT>
__global__ void gemm_bf16_wmma(const float* __restrict__ A,
                               const __bf16* __restrict__ Bpk,
                               __bf16* __restrict__ Out,
                               int M, int Nc) {
    constexpr int KB = K >> 5;          // k-blocks of 32
    const int lane  = threadIdx.x & 31;
    const int wave  = threadIdx.x >> 5;
    const int tilesN = Nc >> 4;
    const int groupsN = tilesN / NT;
    const int tilesM = M >> 4;
    const int unit = blockIdx.x * (blockDim.x >> 5) + wave;
    if (unit >= tilesM * groupsN) return;     // uniform per wave
    const int tm  = unit / groupsN;
    const int tn0 = (unit % groupsN) * NT;

    const int row   = lane & 15;
    const int khalf = lane >> 4;
    const int m0 = tm << 4;

    v8f acc[NT];
    #pragma unroll
    for (int t = 0; t < NT; ++t)
        acc[t] = (v8f){0.f, 0.f, 0.f, 0.f, 0.f, 0.f, 0.f, 0.f};

    const float* arow = A + (size_t)(m0 + row) * K;

    #pragma unroll
    for (int kb = 0; kb < KB; ++kb) {
        const int k0 = kb << 5;
        // A fragment: one contiguous 16-float run (k = 16*khalf + e) -> b128 loads + pk cvt
        const v4f a0 = *(const v4f*)(arow + k0 + (khalf << 4));
        const v4f a1 = *(const v4f*)(arow + k0 + (khalf << 4) + 4);
        const v4f a2 = *(const v4f*)(arow + k0 + (khalf << 4) + 8);
        const v4f a3 = *(const v4f*)(arow + k0 + (khalf << 4) + 12);
        v16bf afrag;
        #pragma unroll
        for (int j = 0; j < 4; ++j) {
            afrag[j]      = (__bf16)a0[j];
            afrag[4 + j]  = (__bf16)a1[j];
            afrag[8 + j]  = (__bf16)a2[j];
            afrag[12 + j] = (__bf16)a3[j];
        }
        #pragma unroll
        for (int t = 0; t < NT; ++t) {
            const v16bf bfrag = *(const v16bf*)(
                Bpk + ((size_t)((tn0 + t) * KB + kb) * 32 + lane) * 16);
            acc[t] = __builtin_amdgcn_wmma_f32_16x16x32_bf16(
                false, afrag, false, bfrag, (short)0, acc[t], false, false);
        }
    }

    // D: VGPR r -> M = r + khalf*8 ; col = lane&15
    #pragma unroll
    for (int t = 0; t < NT; ++t) {
        #pragma unroll
        for (int r = 0; r < 8; ++r) {
            int mrow = r + (khalf << 3);
            Out[(size_t)(m0 + mrow) * Nc + ((tn0 + t) << 4) + row] = (__bf16)acc[t][r];
        }
    }
}

// ---------------- Layer-1 edge kernels (H=8, C=32) ----------------
__global__ void edge_logits1(const __bf16* __restrict__ xl, const __bf16* __restrict__ xr,
                             const float* __restrict__ att, const int* __restrict__ ei,
                             float* __restrict__ logits, float* __restrict__ m) {
    int gid = blockIdx.x * blockDim.x + threadIdx.x;
    int e = gid >> 3;
    int h = gid & 7;
    if (e >= EA) return;
    int src = (e < EE) ? ei[e]      : (e - EE);
    int dst = (e < EE) ? ei[EE + e] : (e - EE);
    const v8bf* pl = (const v8bf*)(xl + (size_t)src * F1 + h * HID);
    const v8bf* pr = (const v8bf*)(xr + (size_t)dst * F1 + h * HID);
    const float* pa = att + h * HID;
    float acc = 0.0f;
    #pragma unroll
    for (int c = 0; c < 4; ++c) {
        v8bf va = pl[c];
        v8bf vb = pr[c];
        #pragma unroll
        for (int j = 0; j < 8; ++j) {
            float v = (float)va[j] + (float)vb[j];
            v = (v > 0.0f) ? v : NEG_SLOPE * v;
            acc += v * pa[c * 8 + j];
        }
    }
    logits[(size_t)e * HEADS + h] = acc;
    atomicMaxFloat(&m[(size_t)dst * HEADS + h], acc);
}

__global__ void edge_expsum1(const int* __restrict__ ei, float* __restrict__ logits,
                             const float* __restrict__ m, float* __restrict__ s) {
    int gid = blockIdx.x * blockDim.x + threadIdx.x;
    int e = gid >> 3;
    int h = gid & 7;
    if (e >= EA) return;
    int dst = (e < EE) ? ei[EE + e] : (e - EE);
    float ex = __expf(logits[(size_t)e * HEADS + h] - m[(size_t)dst * HEADS + h]);
    logits[(size_t)e * HEADS + h] = ex;
    atomicAdd(&s[(size_t)dst * HEADS + h], ex);
}

// one wave per edge; lane owns 8 consecutive features (all within one head)
__global__ void edge_agg1(const __bf16* __restrict__ xl, const float* __restrict__ ex,
                          const float* __restrict__ s, const int* __restrict__ ei,
                          float* __restrict__ outp) {
    int gid  = blockIdx.x * blockDim.x + threadIdx.x;
    int lane = gid & 31;
    int e    = gid >> 5;
    if (e >= EA) return;
    int src = (e < EE) ? ei[e]      : (e - EE);
    int dst = (e < EE) ? ei[EE + e] : (e - EE);
    int h = lane >> 2;   // 8 feats per lane, 32 per head
    float alpha = ex[(size_t)e * HEADS + h] / (s[(size_t)dst * HEADS + h] + 1e-16f);
    int fbase = lane << 3;
    v8bf vl = *(const v8bf*)(xl + (size_t)src * F1 + fbase);
    float* po = outp + (size_t)dst * F1 + fbase;
    #pragma unroll
    for (int j = 0; j < 8; ++j) {
        atomicAdd(&po[j], (float)vl[j] * alpha);
    }
}

// ---------------- Layer-2 edge kernels (H=1, C=32) ----------------
__global__ void edge_logits2(const __bf16* __restrict__ xl, const __bf16* __restrict__ xr,
                             const float* __restrict__ att, const int* __restrict__ ei,
                             float* __restrict__ logits, float* __restrict__ m) {
    int e = blockIdx.x * blockDim.x + threadIdx.x;
    if (e >= EA) return;
    int src = (e < EE) ? ei[e]      : (e - EE);
    int dst = (e < EE) ? ei[EE + e] : (e - EE);
    const v8bf* pl = (const v8bf*)(xl + (size_t)src * HID);
    const v8bf* pr = (const v8bf*)(xr + (size_t)dst * HID);
    float acc = 0.0f;
    #pragma unroll
    for (int c = 0; c < 4; ++c) {
        v8bf va = pl[c];
        v8bf vb = pr[c];
        #pragma unroll
        for (int j = 0; j < 8; ++j) {
            float v = (float)va[j] + (float)vb[j];
            v = (v > 0.0f) ? v : NEG_SLOPE * v;
            acc += v * att[c * 8 + j];
        }
    }
    logits[e] = acc;
    atomicMaxFloat(&m[dst], acc);
}

__global__ void edge_expsum2(const int* __restrict__ ei, float* __restrict__ logits,
                             const float* __restrict__ m, float* __restrict__ s) {
    int e = blockIdx.x * blockDim.x + threadIdx.x;
    if (e >= EA) return;
    int dst = (e < EE) ? ei[EE + e] : (e - EE);
    float ex = __expf(logits[e] - m[dst]);
    logits[e] = ex;
    atomicAdd(&s[dst], ex);
}

__global__ void edge_agg2(const __bf16* __restrict__ xl, const float* __restrict__ ex,
                          const float* __restrict__ s, const int* __restrict__ ei,
                          float* __restrict__ outp) {
    int gid = blockIdx.x * blockDim.x + threadIdx.x;
    int c = gid & 31;
    int e = gid >> 5;
    if (e >= EA) return;
    int src = (e < EE) ? ei[e]      : (e - EE);
    int dst = (e < EE) ? ei[EE + e] : (e - EE);
    float alpha = ex[e] / (s[dst] + 1e-16f);
    atomicAdd(&outp[(size_t)dst * HID + c], (float)xl[(size_t)src * HID + c] * alpha);
}

// ---------------- Bias + ELU (in place); F must be a power of two ----------------
__global__ void bias_elu(float* __restrict__ buf, const float* __restrict__ bias,
                         int F, int total) {
    int i = blockIdx.x * blockDim.x + threadIdx.x;
    if (i >= total) return;
    float v = buf[i] + bias[i & (F - 1)];
    buf[i] = (v > 0.0f) ? v : (__expf(v) - 1.0f);
}

// ---------------- Final linear 32 -> 40 ----------------
__global__ void final_linear(const float* __restrict__ h2, const float* __restrict__ w,
                             const float* __restrict__ b, float* __restrict__ out) {
    int gid = blockIdx.x * blockDim.x + threadIdx.x;
    if (gid >= NN * OUT_DIM) return;
    int o = gid % OUT_DIM;
    int n = gid / OUT_DIM;
    const v4f* ph = (const v4f*)(h2 + (size_t)n * HID);
    float acc = b[o];
    #pragma unroll
    for (int c4 = 0; c4 < 8; ++c4) {
        v4f hv = ph[c4];
        #pragma unroll
        for (int j = 0; j < 4; ++j)
            acc += hv[j] * w[(c4 * 4 + j) * OUT_DIM + o];
    }
    out[gid] = acc;
}

// ---------------- Host launcher ----------------
static inline int cdiv(long long a, long long b) { return (int)((a + b - 1) / b); }

extern "C" void kernel_launch(void* const* d_in, const int* in_sizes, int n_in,
                              void* d_out, int out_size, void* d_ws, size_t ws_size,
                              hipStream_t stream) {
    const float* x     = (const float*)d_in[0];
    const int*   ei    = (const int*)d_in[1];          // [2, E] flat, int32
    const float* w1l   = (const float*)d_in[2];
    const float* w1r   = (const float*)d_in[3];
    const float* att1  = (const float*)d_in[4];
    const float* b1    = (const float*)d_in[5];
    const float* w2l   = (const float*)d_in[6];
    const float* w2r   = (const float*)d_in[7];
    const float* att2  = (const float*)d_in[8];
    const float* b2    = (const float*)d_in[9];
    const float* wlin  = (const float*)d_in[10];
    const float* blin  = (const float*)d_in[11];
    float* out = (float*)d_out;

    // Workspace carve-out (256B aligned)
    char* base = (char*)d_ws;
    size_t off = 0;
    auto carve = [&](size_t bytes) -> char* {
        char* p = base + off;
        off += (bytes + 255) & ~(size_t)255;
        return p;
    };
    __bf16* xl1   = (__bf16*)carve((size_t)NN * F1 * 2);
    __bf16* xr1   = (__bf16*)carve((size_t)NN * F1 * 2);
    float*  lg1   = (float*) carve((size_t)EA * HEADS * 4);
    float*  m1    = (float*) carve((size_t)NN * HEADS * 4);
    float*  s1    = (float*) carve((size_t)NN * HEADS * 4);
    float*  h1    = (float*) carve((size_t)NN * F1 * 4);
    __bf16* xl2   = (__bf16*)carve((size_t)NN * HID * 2);
    __bf16* xr2   = (__bf16*)carve((size_t)NN * HID * 2);
    float*  lg2   = (float*) carve((size_t)EA * 4);
    float*  m2    = (float*) carve((size_t)NN * 4);
    float*  s2    = (float*) carve((size_t)NN * 4);
    float*  h2    = (float*) carve((size_t)NN * HID * 4);
    __bf16* b1lp  = (__bf16*)carve((size_t)IN_DIM * F1 * 2);  // packed fragments
    __bf16* b1rp  = (__bf16*)carve((size_t)IN_DIM * F1 * 2);
    __bf16* b2lp  = (__bf16*)carve((size_t)F1 * HID * 2);
    __bf16* b2rp  = (__bf16*)carve((size_t)F1 * HID * 2);

    const int T = 256;

    // Pack weights into WMMA B-fragment layout (bf16)
    {
        int tot1 = (F1 / 16) * (IN_DIM / 32) * 32;   // 2048
        int tot2 = (HID / 16) * (F1 / 32) * 32;      // 512
        pack_b_frag<<<cdiv(tot1, T), T, 0, stream>>>(w1l, b1lp, IN_DIM, F1);
        pack_b_frag<<<cdiv(tot1, T), T, 0, stream>>>(w1r, b1rp, IN_DIM, F1);
        pack_b_frag<<<cdiv(tot2, T), T, 0, stream>>>(w2l, b2lp, F1, HID);
        pack_b_frag<<<cdiv(tot2, T), T, 0, stream>>>(w2r, b2rp, F1, HID);
    }

    // Layer-1 linear transforms: [N,128] @ [128,256] via WMMA bf16 (2 N-tiles/wave)
    {
        int units = (NN / 16) * ((F1 / 16) / 2);   // 3125 * 8
        int blocks = cdiv(units, T / 32);
        gemm_bf16_wmma<IN_DIM, 2><<<blocks, T, 0, stream>>>(x, b1lp, xl1, NN, F1);
        gemm_bf16_wmma<IN_DIM, 2><<<blocks, T, 0, stream>>>(x, b1rp, xr1, NN, F1);
    }

    // Init segment buffers
    fill_f32<<<cdiv(NN * HEADS, T), T, 0, stream>>>(m1, -INFINITY, NN * HEADS);
    fill_f32<<<cdiv(NN * HEADS, T), T, 0, stream>>>(s1, 0.0f, NN * HEADS);
    fill_f32<<<cdiv(NN * F1, T), T, 0, stream>>>(h1, 0.0f, NN * F1);

    // Layer-1 edge phase
    edge_logits1<<<cdiv((long long)EA * HEADS, T), T, 0, stream>>>(xl1, xr1, att1, ei, lg1, m1);
    edge_expsum1<<<cdiv((long long)EA * HEADS, T), T, 0, stream>>>(ei, lg1, m1, s1);
    edge_agg1<<<cdiv((long long)EA * 32, T), T, 0, stream>>>(xl1, lg1, s1, ei, h1);
    bias_elu<<<cdiv((long long)NN * F1, T), T, 0, stream>>>(h1, b1, F1, NN * F1);

    // Layer-2 linear transforms: [N,256] @ [256,32] (one wave covers both N-tiles)
    {
        int units = (NN / 16) * ((HID / 16) / 2);  // 3125
        int blocks = cdiv(units, T / 32);
        gemm_bf16_wmma<F1, 2><<<blocks, T, 0, stream>>>(h1, b2lp, xl2, NN, HID);
        gemm_bf16_wmma<F1, 2><<<blocks, T, 0, stream>>>(h1, b2rp, xr2, NN, HID);
    }

    fill_f32<<<cdiv(NN, T), T, 0, stream>>>(m2, -INFINITY, NN);
    fill_f32<<<cdiv(NN, T), T, 0, stream>>>(s2, 0.0f, NN);
    fill_f32<<<cdiv(NN * HID, T), T, 0, stream>>>(h2, 0.0f, NN * HID);

    // Layer-2 edge phase
    edge_logits2<<<cdiv(EA, T), T, 0, stream>>>(xl2, xr2, att2, ei, lg2, m2);
    edge_expsum2<<<cdiv(EA, T), T, 0, stream>>>(ei, lg2, m2, s2);
    edge_agg2<<<cdiv((long long)EA * 32, T), T, 0, stream>>>(xl2, lg2, s2, ei, h2);
    bias_elu<<<cdiv((long long)NN * HID, T), T, 0, stream>>>(h2, b2, HID, NN * HID);

    // Final linear to d_out
    final_linear<<<cdiv((long long)NN * OUT_DIM, T), T, 0, stream>>>(h2, wlin, blin, out);

    (void)in_sizes; (void)n_in; (void)out_size; (void)ws_size;
}
